// RoIAggPool3d_46084999086894
// MI455X (gfx1250) — matrix operand order repaired and checked by reading, not compiled, not run on hardware
//
#include <hip/hip_runtime.h>
#include <hip/hip_bf16.h>

#define BB 2
#define NN 2048
#define RR 128
#define CINV 256
#define NS 16
#define OUTV 5
#define RADIUSF 0.3f
#define EPSV 1e-5f
#define K1P 288   // padded K for layer-1 GEMM (256 feats + 3 coords + 29 zero pad)

typedef __attribute__((ext_vector_type(16))) _Float16 v16h;
typedef __attribute__((ext_vector_type(8)))  _Float16 v8h;
typedef __attribute__((ext_vector_type(8)))  float    v8f;
typedef __attribute__((ext_vector_type(4)))  float    v4f;

// -------------------------------------------------------------------------
// prep: features (B,CIN,N) f32 -> featT (B,N,CIN) f16 (contiguous gather rows)
__global__ __launch_bounds__(256) void k_transpose_feats(const float* __restrict__ f,
                                                         _Float16* __restrict__ ft){
  int tid = blockIdx.x * 256 + threadIdx.x;            // B*CIN*N
  if (tid >= BB * CINV * NN) return;
  int n = tid & (NN - 1);
  int c = (tid >> 11) & (CINV - 1);
  int b = tid >> 19;
  ft[((size_t)b * NN + n) * CINV + c] = (_Float16)f[tid];
}

// W_mlp (256x259) f32 -> Wh (256x288) f16, K reordered: [features(256) | gx(3) | pad(29)]
__global__ __launch_bounds__(256) void k_conv_wmlp(const float* __restrict__ W,
                                                   _Float16* __restrict__ Wh){
  int tid = blockIdx.x * 256 + threadIdx.x;            // 256*288
  if (tid >= 256 * K1P) return;
  int k = tid % K1P, o = tid / K1P;
  float v = 0.f;
  if (k < 256)      v = W[o * 259 + 3 + k];            // feature channels
  else if (k < 259) v = W[o * 259 + (k - 256)];        // gx coords
  Wh[tid] = (_Float16)v;
}

__global__ __launch_bounds__(256) void k_conv_w(const float* __restrict__ W,
                                                _Float16* __restrict__ Wh, int n){
  int tid = blockIdx.x * 256 + threadIdx.x;
  if (tid < n) Wh[tid] = (_Float16)W[tid];
}

// -------------------------------------------------------------------------
// ball query: first 16 smallest-index neighbors within RADIUS; pad with first hit.
__global__ __launch_bounds__(256) void k_ballq(const float* __restrict__ pts,
                                               int* __restrict__ idxO,
                                               _Float16* __restrict__ gxh){
  __shared__ float sx[NN], sy[NN], sz[NN];
  int b = blockIdx.x >> 3;                             // 8 blocks of 256 per batch
  int nbase = (blockIdx.x & 7) * 256;
  const float* P = pts + (size_t)b * NN * 3;
  for (int i = threadIdx.x; i < NN; i += 256){
    sx[i] = P[i * 3 + 0]; sy[i] = P[i * 3 + 1]; sz[i] = P[i * 3 + 2];
  }
  __syncthreads();
  int n = nbase + threadIdx.x;
  int pt = b * NN + n;
  float cx = sx[n], cy = sy[n], cz = sz[n];
  const float r2 = RADIUSF * RADIUSF;
  const float inv_r = 1.f / RADIUSF;
  int cnt = 0, firstj = 0;
  float fgx = 0.f, fgy = 0.f, fgz = 0.f;
  for (int j = 0; j < NN; j++){
    float dx = sx[j] - cx, dy = sy[j] - cy, dz = sz[j] - cz;
    if (dx * dx + dy * dy + dz * dz < r2){
      idxO[pt * NS + cnt] = j;
      _Float16* g = gxh + ((size_t)pt * NS + cnt) * 4;
      g[0] = (_Float16)(dx * inv_r); g[1] = (_Float16)(dy * inv_r);
      g[2] = (_Float16)(dz * inv_r); g[3] = (_Float16)0.f;
      if (cnt == 0){ firstj = j; fgx = dx; fgy = dy; fgz = dz; }
      cnt++;
      if (cnt == NS) break;
    }
  }
  for (int s = cnt; s < NS; s++){                      // pad with first hit
    idxO[pt * NS + s] = firstj;
    _Float16* g = gxh + ((size_t)pt * NS + s) * 4;
    g[0] = (_Float16)(fgx * inv_r); g[1] = (_Float16)(fgy * inv_r);
    g[2] = (_Float16)(fgz * inv_r); g[3] = (_Float16)0.f;
  }
}

// -------------------------------------------------------------------------
// Layer-1 GEMM: 4 points per block (64 sample-columns x 288 K), 256 out channels.
// Block = 4 waves. Per wave: per out-tile, preload 9 A-fragments to registers,
// reuse across 4 column-tiles (4x arithmetic intensity on global A path).
// Feature gather staged to LDS via CDNA5 async global->LDS (ASYNCcnt) copies.
// Epilogue: C tile staged to per-wave LDS scratch; lanes 0-15 reduce sum/sumsq,
// lanes 16-31 reduce max/min per channel row (for fused BN+ReLU+max-over-s).
__global__ __launch_bounds__(128) void k_gemm1(const _Float16* __restrict__ Wh,
                                               const _Float16* __restrict__ featT,
                                               const int* __restrict__ idx,
                                               const _Float16* __restrict__ gxh,
                                               float* __restrict__ psum,
                                               float* __restrict__ psumsq,
                                               float* __restrict__ hmax,
                                               float* __restrict__ hmin){
  __shared__ __attribute__((aligned(16))) _Float16 sB[64 * K1P];   // 36,864 B
  __shared__ __attribute__((aligned(16))) float sC[4 * 256];       //  4,096 B
  int pt0 = blockIdx.x * 4;                            // 4 points, same batch
  int b = pt0 >> 11;
  int tid = threadIdx.x;
  {
    int col = tid >> 1, sub = tid & 1;                 // 64 columns, 2 threads each
    int p = pt0 + (col >> 4), s = col & 15;
    int nb = idx[p * NS + s];
    const _Float16* src = featT + ((size_t)b * NN + nb) * CINV;
    unsigned dstoff = (unsigned)(size_t)&sB[col * K1P];
    #pragma unroll
    for (int i = 0; i < 16; i++){
      int off = (i * 2 + sub) * 8;                     // 8-half (16B) chunks
      unsigned l = dstoff + off * 2;
      const _Float16* g = src + off;
      asm volatile("global_load_async_to_lds_b128 %0, %1, off"
                   :: "v"(l), "v"(g) : "memory");
    }
    if (sub == 0){
      _Float16* dst = &sB[col * K1P];
      const _Float16* gx = gxh + ((size_t)p * NS + s) * 4;
      dst[256] = gx[0]; dst[257] = gx[1]; dst[258] = gx[2];
      #pragma unroll
      for (int k = 259; k < K1P; k++) dst[k] = (_Float16)0.f;
    }
  }
  asm volatile("s_wait_asynccnt 0x0" ::: "memory");
  __syncthreads();

  int wave = tid >> 5, lane = tid & 31;
  int half = lane >> 4, lr = lane & 15;
  float* sc = &sC[wave * 256];
  for (int ot = 0; ot < 4; ot++){
    int o0 = (wave * 4 + ot) * 16;
    // preload all 9 A fragments (rows o0..o0+15) into registers
    const _Float16* Arow = Wh + (size_t)(o0 + lr) * K1P + half * 8;
    v16h A[9];
    #pragma unroll
    for (int t = 0; t < 9; t++){
      v8h a0 = *(const v8h*)(Arow + t * 32);
      v8h a1 = *(const v8h*)(Arow + t * 32 + 16);
      #pragma unroll
      for (int e = 0; e < 8; e++){ A[t][e] = a0[e]; A[t][e + 8] = a1[e]; }
    }
    for (int ct = 0; ct < 4; ct++){
      v8f acc = {};
      const _Float16* Bcol = &sB[(ct * 16 + lr) * K1P + half * 8];
      #pragma unroll
      for (int t = 0; t < 9; t++){
        v8h b0 = *(const v8h*)(Bcol + t * 32);
        v8h b1 = *(const v8h*)(Bcol + t * 32 + 16);
        v16h bv;
        #pragma unroll
        for (int e = 0; e < 8; e++){ bv[e] = b0[e]; bv[e + 8] = b1[e]; }
        acc = __builtin_amdgcn_wmma_f32_16x16x32_f16(false, A[t], false, bv,
                                                     (short)0, acc, false, false);
      }
      // ---- epilogue: stage C tile, per-lane row reductions (DS in-order) ----
      #pragma unroll
      for (int r = 0; r < 8; r++) sc[(half * 8 + r) * 16 + lr] = acc[r];
      v4f x0 = *(const v4f*)(sc + lr * 16);
      v4f x1 = *(const v4f*)(sc + lr * 16 + 4);
      v4f x2 = *(const v4f*)(sc + lr * 16 + 8);
      v4f x3 = *(const v4f*)(sc + lr * 16 + 12);
      size_t o = (size_t)(pt0 + ct) * 256 + o0 + lr;
      if (half == 0){
        float s = 0.f, q = 0.f;
        #pragma unroll
        for (int e = 0; e < 4; e++){
          s += x0[e] + x1[e] + x2[e] + x3[e];
          q += x0[e]*x0[e] + x1[e]*x1[e] + x2[e]*x2[e] + x3[e]*x3[e];
        }
        psum[o] = s; psumsq[o] = q;
      } else {
        float mx = x0[0], mn = x0[0];
        #pragma unroll
        for (int e = 0; e < 4; e++){
          mx = fmaxf(mx, fmaxf(fmaxf(x0[e], x1[e]), fmaxf(x2[e], x3[e])));
          mn = fminf(mn, fminf(fminf(x0[e], x1[e]), fminf(x2[e], x3[e])));
        }
        hmax[o] = mx; hmin[o] = mn;
      }
    }
  }
}

// BN stats for layer 1: reduce per-point partials over 4096 points; count = B*N*S.
__global__ __launch_bounds__(256) void k_stats1(const float* __restrict__ psum,
                                                const float* __restrict__ psumsq,
                                                const float* __restrict__ g,
                                                const float* __restrict__ bt,
                                                float* __restrict__ abn){
  int ch = blockIdx.x, t = threadIdx.x;
  float s = 0.f, q = 0.f;
  for (int pt = t; pt < BB * NN; pt += 256){
    s += psum[(size_t)pt * 256 + ch];
    q += psumsq[(size_t)pt * 256 + ch];
  }
  __shared__ float ss[256], sq[256];
  ss[t] = s; sq[t] = q; __syncthreads();
  for (int o = 128; o > 0; o >>= 1){
    if (t < o){ ss[t] += ss[t + o]; sq[t] += sq[t + o]; }
    __syncthreads();
  }
  if (t == 0){
    float cnt = (float)(BB * NN * NS);
    float m = ss[0] / cnt;
    float var = sq[0] / cnt - m * m;
    float a = g[ch] * rsqrtf(var + EPSV);
    abn[ch] = a; abn[256 + ch] = bt[ch] - m * a;
  }
}

// fused BN + ReLU + max-over-samples: max_s relu(a*h+c) = relu(a*(a>=0?hmax:hmin)+c)
__global__ __launch_bounds__(256) void k_apply1(const float* __restrict__ hmax,
                                                const float* __restrict__ hmin,
                                                const float* __restrict__ abn,
                                                _Float16* __restrict__ Xout){
  int tid = blockIdx.x * 256 + threadIdx.x;            // B*N*256
  if (tid >= BB * NN * 256) return;
  int ch = tid & 255;
  float a = abn[ch], c = abn[256 + ch];
  float v = a * (a >= 0.f ? hmax[tid] : hmin[tid]) + c;
  Xout[tid] = (_Float16)fmaxf(v, 0.f);
}

// -------------------------------------------------------------------------
// FC GEMM (layers 2/3): 128 outs, KD in {256,128}, cols = B*N (16 per block).
__global__ __launch_bounds__(128) void k_gemm_fc(const _Float16* __restrict__ Wk,
                                                 const _Float16* __restrict__ X,
                                                 const float* __restrict__ bias,
                                                 float* __restrict__ H, int KD){
  __shared__ __attribute__((aligned(16))) _Float16 sB[16 * 256];
  int blk = blockIdx.x, tid = threadIdx.x;
  {
    int col = tid >> 3, sub = tid & 7;
    const _Float16* src = X + (size_t)(blk * 16 + col) * KD;
    _Float16* dst = &sB[col * KD];
    for (int off = sub * 8; off < KD; off += 64)
      *(v8h*)(dst + off) = *(const v8h*)(src + off);
  }
  __syncthreads();
  int wave = tid >> 5, lane = tid & 31, half = lane >> 4, lr = lane & 15;
  int nk = KD >> 5;
  for (int ot = 0; ot < 2; ot++){
    int o0 = (wave * 2 + ot) * 16;
    v8f acc = {};
    const _Float16* Arow = Wk + (size_t)(o0 + lr) * KD + half * 8;
    const _Float16* Bcol = &sB[lr * KD + half * 8];
    for (int t = 0; t < nk; t++){
      v8h a0 = *(const v8h*)(Arow + t * 32);
      v8h a1 = *(const v8h*)(Arow + t * 32 + 16);
      v8h b0 = *(const v8h*)(Bcol + t * 32);
      v8h b1 = *(const v8h*)(Bcol + t * 32 + 16);
      v16h av, bv;
      #pragma unroll
      for (int e = 0; e < 8; e++){ av[e] = a0[e]; av[e + 8] = a1[e];
                                   bv[e] = b0[e]; bv[e + 8] = b1[e]; }
      acc = __builtin_amdgcn_wmma_f32_16x16x32_f16(false, av, false, bv,
                                                   (short)0, acc, false, false);
    }
    int gcol = blk * 16 + lr;
    #pragma unroll
    for (int r = 0; r < 8; r++){
      int row = o0 + half * 8 + r;
      H[(size_t)gcol * 128 + row] = acc[r] + bias[row];
    }
  }
}

// BN stats for FC layers: reduce H[col*128+ch] over 4096 columns.
__global__ __launch_bounds__(256) void k_stats_fc(const float* __restrict__ H,
                                                  const float* __restrict__ g,
                                                  const float* __restrict__ bt,
                                                  float* __restrict__ abn){
  int ch = blockIdx.x, t = threadIdx.x;
  float s = 0.f, q = 0.f;
  for (int col = t; col < BB * NN; col += 256){
    float v = H[(size_t)col * 128 + ch];
    s += v; q += v * v;
  }
  __shared__ float ss[256], sq[256];
  ss[t] = s; sq[t] = q; __syncthreads();
  for (int o = 128; o > 0; o >>= 1){
    if (t < o){ ss[t] += ss[t + o]; sq[t] += sq[t + o]; }
    __syncthreads();
  }
  if (t == 0){
    float cnt = (float)(BB * NN);
    float m = ss[0] / cnt;
    float var = sq[0] / cnt - m * m;
    float a = g[ch] * rsqrtf(var + EPSV);
    abn[ch] = a; abn[128 + ch] = bt[ch] - m * a;
  }
}

__global__ __launch_bounds__(256) void k_apply_fc(const float* __restrict__ H,
                                                  const float* __restrict__ abn,
                                                  _Float16* __restrict__ Xout){
  int tid = blockIdx.x * 256 + threadIdx.x;            // B*N*128
  if (tid >= BB * NN * 128) return;
  int ch = tid & 127;
  Xout[tid] = (_Float16)fmaxf(abn[ch] * H[tid] + abn[128 + ch], 0.f);
}

// layer-3 apply: writes feats output (B,128,N) f32 + featsT (col-major) for pooling
__global__ __launch_bounds__(256) void k_apply3(const float* __restrict__ H,
                                                const float* __restrict__ abn,
                                                float* __restrict__ outFeats,
                                                float* __restrict__ featsT){
  int tid = blockIdx.x * 256 + threadIdx.x;            // B*N*128
  if (tid >= BB * NN * 128) return;
  int ch = tid & 127, col = tid >> 7;
  float v = fmaxf(abn[ch] * H[tid] + abn[128 + ch], 0.f);
  featsT[tid] = v;
  int b = col >> 11, n = col & (NN - 1);
  outFeats[((size_t)b * 128 + ch) * NN + n] = v;
}

// -------------------------------------------------------------------------
// RoI voxelization: per (b,r,n) compute 5^3 voxel id or -1.
__global__ __launch_bounds__(256) void k_voxel(const float* __restrict__ pts,
                                               const float* __restrict__ rois,
                                               signed char* __restrict__ vm){
  int tid = blockIdx.x * 256 + threadIdx.x;            // B*R*N
  if (tid >= BB * RR * NN) return;
  int n = tid & (NN - 1);
  int r = (tid >> 11) & (RR - 1);
  int b = tid >> 18;
  const float* roi = rois + ((size_t)b * RR + r) * 7;
  const float* p = pts + ((size_t)b * NN + n) * 3;
  float rx = p[0] - roi[0], ryv = p[1] - roi[1], rz = p[2] - roi[2];
  float cs = cosf(roi[6]), sn = sinf(roi[6]);
  float lx =  rx * cs + ryv * sn;
  float ly = -rx * sn + ryv * cs;
  float lz =  rz;
  float hx = roi[3] * 0.5f, hy = roi[4] * 0.5f, hz = roi[5] * 0.5f;
  signed char out = -1;
  if (fabsf(lx) < hx && fabsf(ly) < hy && fabsf(lz) < hz){
    int vx = (int)floorf((lx + hx) / (2.f * hx) * (float)OUTV);
    int vy = (int)floorf((ly + hy) / (2.f * hy) * (float)OUTV);
    int vz = (int)floorf((lz + hz) / (2.f * hz) * (float)OUTV);
    vx = vx < 0 ? 0 : (vx > OUTV - 1 ? OUTV - 1 : vx);
    vy = vy < 0 ? 0 : (vy > OUTV - 1 ? OUTV - 1 : vy);
    vz = vz < 0 ? 0 : (vz > OUTV - 1 ? OUTV - 1 : vz);
    out = (signed char)((vx * OUTV + vy) * OUTV + vz);
  }
  vm[tid] = out;
}

__global__ __launch_bounds__(256) void k_zero(float* __restrict__ p, int n){
  int i = blockIdx.x * 256 + threadIdx.x;
  if (i < n) p[i] = 0.f;
}

// segment max via u32 atomicMax (features >= 0 post-ReLU; 0-init == ref's -inf->0)
__global__ __launch_bounds__(128) void k_pool(const signed char* __restrict__ vm,
                                              const float* __restrict__ featsT,
                                              float* __restrict__ pooled){
  int br = blockIdx.x;                                 // B*R
  int c = threadIdx.x;                                 // 128 channels
  unsigned* P = (unsigned*)(pooled + (size_t)br * 125 * 128);
  const signed char* V = vm + (size_t)br * NN;
  int b = br >> 7;
  for (int n = 0; n < NN; n++){
    int v = V[n];
    if (v >= 0){
      float val = featsT[((size_t)b * NN + n) * 128 + c];
      atomicMax(&P[v * 128 + c], __float_as_uint(val));
    }
  }
}

// -------------------------------------------------------------------------
extern "C" void kernel_launch(void* const* d_in, const int* in_sizes, int n_in,
                              void* d_out, int out_size, void* d_ws, size_t ws_size,
                              hipStream_t stream){
  const float* points   = (const float*)d_in[0];
  const float* features = (const float*)d_in[1];
  const float* rois     = (const float*)d_in[2];
  const float* W_mlp    = (const float*)d_in[3];
  const float* g_mlp    = (const float*)d_in[4];
  const float* bt_mlp   = (const float*)d_in[5];
  const float* W1       = (const float*)d_in[6];
  const float* b1       = (const float*)d_in[7];
  const float* g1       = (const float*)d_in[8];
  const float* bt1      = (const float*)d_in[9];
  const float* W2       = (const float*)d_in[10];
  const float* b2       = (const float*)d_in[11];
  const float* g2       = (const float*)d_in[12];
  const float* bt2      = (const float*)d_in[13];

  char* ws = (char*)d_ws;
  size_t off = 0;
  auto alloc = [&](size_t bytes) -> char* {
    char* p = ws + off;
    off += (bytes + 255) & ~(size_t)255;
    return p;
  };
  const int NPT = BB * NN;                             // 4096 points/columns
  _Float16* featT   = (_Float16*)alloc((size_t)NPT * CINV * 2);
  _Float16* Wh      = (_Float16*)alloc((size_t)256 * K1P * 2);
  _Float16* W1h     = (_Float16*)alloc((size_t)128 * 256 * 2);
  _Float16* W2h     = (_Float16*)alloc((size_t)128 * 128 * 2);
  int*      idx     = (int*)     alloc((size_t)NPT * NS * 4);
  _Float16* gxh     = (_Float16*)alloc((size_t)NPT * NS * 4 * 2);
  float*    psum    = (float*)   alloc((size_t)NPT * 256 * 4);
  float*    psumsq  = (float*)   alloc((size_t)NPT * 256 * 4);
  float*    hmax    = (float*)   alloc((size_t)NPT * 256 * 4);
  float*    hmin    = (float*)   alloc((size_t)NPT * 256 * 4);
  float*    abn1    = (float*)   alloc((size_t)512 * 4);
  _Float16* feats1h = (_Float16*)alloc((size_t)NPT * 256 * 2);
  float*    h2      = (float*)   alloc((size_t)NPT * 128 * 4);
  float*    abn2    = (float*)   alloc((size_t)256 * 4);
  _Float16* feats2h = (_Float16*)alloc((size_t)NPT * 128 * 2);
  float*    h3      = (float*)   alloc((size_t)NPT * 128 * 4);
  float*    abn3    = (float*)   alloc((size_t)256 * 4);
  float*    featsT  = (float*)   alloc((size_t)NPT * 128 * 4);
  signed char* vm   = (signed char*)alloc((size_t)BB * RR * NN);

  float* outFeats = (float*)d_out;                     // (B,128,N)
  float* pooled   = outFeats + (size_t)BB * 128 * NN;  // (B,R,5,5,5,128)

  // prep
  k_transpose_feats<<<(BB * CINV * NN) / 256, 256, 0, stream>>>(features, featT);
  k_conv_wmlp<<<(256 * K1P) / 256, 256, 0, stream>>>(W_mlp, Wh);
  k_conv_w<<<(128 * 256) / 256, 256, 0, stream>>>(W1, W1h, 128 * 256);
  k_conv_w<<<(128 * 128) / 256, 256, 0, stream>>>(W2, W2h, 128 * 128);
  k_ballq<<<BB * (NN / 256), 256, 0, stream>>>(points, idx, gxh);

  // layer 1: WMMA GEMM + fused stats/max epilogue
  k_gemm1<<<NPT / 4, 128, 0, stream>>>(Wh, featT, idx, gxh, psum, psumsq, hmax, hmin);
  k_stats1<<<256, 256, 0, stream>>>(psum, psumsq, g_mlp, bt_mlp, abn1);
  k_apply1<<<(NPT * 256) / 256, 256, 0, stream>>>(hmax, hmin, abn1, feats1h);

  // layer 2
  k_gemm_fc<<<NPT / 16, 128, 0, stream>>>(W1h, feats1h, b1, h2, 256);
  k_stats_fc<<<128, 256, 0, stream>>>(h2, g1, bt1, abn2);
  k_apply_fc<<<(NPT * 128) / 256, 256, 0, stream>>>(h2, abn2, feats2h);

  // layer 3
  k_gemm_fc<<<NPT / 16, 128, 0, stream>>>(W2h, feats2h, b2, h3, 128);
  k_stats_fc<<<128, 256, 0, stream>>>(h3, g2, bt2, abn3);
  k_apply3<<<(NPT * 128) / 256, 256, 0, stream>>>(h3, abn3, outFeats, featsT);

  // RoI voxel max-pool
  k_voxel<<<(BB * RR * NN) / 256, 256, 0, stream>>>(points, rois, vm);
  int npool = BB * RR * 125 * 128;
  k_zero<<<(npool + 255) / 256, 256, 0, stream>>>(pooled, npool);
  k_pool<<<BB * RR, 128, 0, stream>>>(vm, featsT, pooled);
}
